// TransGNN_68547678044259
// MI455X (gfx1250) — matrix operand order repaired and checked by reading, not compiled, not run on hardware
//
#include <hip/hip_runtime.h>
#include <math.h>

typedef __attribute__((ext_vector_type(16))) _Float16 v16h;
typedef __attribute__((ext_vector_type(8)))  float    v8f;

// ---------- monotonic float <-> unsigned key (for atomicMax on floats) ----------
__device__ __forceinline__ unsigned fkey(float f) {
    unsigned u = __float_as_uint(f);
    return (u & 0x80000000u) ? ~u : (u | 0x80000000u);
}
__device__ __forceinline__ float funkey(unsigned k) {
    unsigned u = (k & 0x80000000u) ? (k & 0x7FFFFFFFu) : ~k;
    return __uint_as_float(u);
}

// ---------------------------------------------------------------------------
// WMMA GEMM: Dst[M,Nc] = act(A[M,K] @ W[K,Nc] + bias (+ resid))
// fp32 in/out; split-f16 (hi+lo) => 3 x v_wmma_f32_16x16x32_f16 per K-step
// One wave computes one 16x16 output tile. M must be a multiple of 16.
// Full K-steps run in a branch-free loop; the (<32) tail runs exactly once.
// ---------------------------------------------------------------------------
__global__ __launch_bounds__(128)
void wmma_gemm(const float* __restrict__ A, const float* __restrict__ W,
               const float* __restrict__ bias, const float* __restrict__ resid,
               float* __restrict__ Dst, int M, int K, int Nc, int do_relu)
{
    const int lane  = threadIdx.x;                       // 0..31
    const int mtile = blockIdx.y * blockDim.y + threadIdx.y;
    if (mtile * 16 >= M) return;                         // wave-uniform exit
    const int m0 = mtile * 16;
    const int n0 = blockIdx.x * 16;
    const int half = lane >> 4;                          // 0 / 1
    const int l15  = lane & 15;
    const int m = m0 + l15;                              // A row owned by lane
    const int n = n0 + l15;                              // B col owned by lane

    v8f c = {0.f, 0.f, 0.f, 0.f, 0.f, 0.f, 0.f, 0.f};

    const int kfull = K & ~31;                           // uniform

    // ---------------- branch-free full K-steps (hot path) ----------------
    for (int kc = 0; kc < kfull; kc += 32) {
        v16h ah, al, bh, bl;
        const float* Ar = A + (size_t)m * K + kc;
        const float* Wb = W + (size_t)kc * Nc + n;
        // A halves j<8  -> K = 8*half + j        (8 contiguous floats)
        // A halves j>=8 -> K = 16 + 8*half + j-8 (8 contiguous floats)
        const float* Ag0 = Ar + 8 * half;
        const float* Ag1 = Ar + 16 + 8 * half;
        float4 a0 = *(const float4*)(Ag0);
        float4 a1 = *(const float4*)(Ag0 + 4);
        float4 a2 = *(const float4*)(Ag1);
        float4 a3 = *(const float4*)(Ag1 + 4);
        float av[16] = { a0.x, a0.y, a0.z, a0.w, a1.x, a1.y, a1.z, a1.w,
                         a2.x, a2.y, a2.z, a2.w, a3.x, a3.y, a3.z, a3.w };
#pragma unroll
        for (int j = 0; j < 16; ++j) {
            // B halves j -> K = 8*half + j (j<8), 16 + 8*half + j-8 (j>=8) (coalesced across lanes per j)
            float bv = Wb[(size_t)((j < 8) ? (8 * half + j) : (8 + 8 * half + j)) * Nc];
            _Float16 avh = (_Float16)av[j];
            _Float16 bvh = (_Float16)bv;
            ah[j] = avh;  al[j] = (_Float16)(av[j] - (float)avh);
            bh[j] = bvh;  bl[j] = (_Float16)(bv - (float)bvh);
        }
        c = __builtin_amdgcn_wmma_f32_16x16x32_f16(false, ah, false, bh, (short)0, c, false, false);
        c = __builtin_amdgcn_wmma_f32_16x16x32_f16(false, ah, false, bl, (short)0, c, false, false);
        c = __builtin_amdgcn_wmma_f32_16x16x32_f16(false, al, false, bh, (short)0, c, false, false);
    }

    // ---------------- single guarded tail (only K=6 / K=4 layers) ----------------
    if (kfull < K) {
        v16h ah, al, bh, bl;
        const int kc = kfull;
        const float* Ar = A + (size_t)m * K + kc;
        const float* Wb = W + (size_t)kc * Nc + n;
#pragma unroll
        for (int j = 0; j < 16; ++j) {
            const int ka = (j < 8) ? (8 * half + j) : (8 + 8 * half + j);
            const int kb = (j < 8) ? (8 * half + j) : (8 + 8 * half + j);
            float av = (kc + ka < K) ? Ar[ka] : 0.0f;
            float bv = (kc + kb < K) ? Wb[(size_t)kb * Nc] : 0.0f;
            _Float16 avh = (_Float16)av;
            _Float16 bvh = (_Float16)bv;
            ah[j] = avh;  al[j] = (_Float16)(av - (float)avh);
            bh[j] = bvh;  bl[j] = (_Float16)(bv - (float)bvh);
        }
        c = __builtin_amdgcn_wmma_f32_16x16x32_f16(false, ah, false, bh, (short)0, c, false, false);
        c = __builtin_amdgcn_wmma_f32_16x16x32_f16(false, ah, false, bl, (short)0, c, false, false);
        c = __builtin_amdgcn_wmma_f32_16x16x32_f16(false, al, false, bh, (short)0, c, false, false);
    }

    const float bn = bias ? bias[n] : 0.0f;
#pragma unroll
    for (int r = 0; r < 8; ++r) {
        const int mm = m0 + 8 * half + r;                // C/D layout: row = r + 8*half
        float val = c[r] + bn;
        if (resid)   val += resid[(size_t)mm * Nc + n];
        if (do_relu) val  = fmaxf(val, 0.0f);
        Dst[(size_t)mm * Nc + n] = val;
    }
}

// ---------------- segment_sum of (deduplicated) edge features -----------------
// recv = concat(ei1, ei0); edge row e2 uses e_half[e2 % E]
__global__ void scatter_agg(const float* __restrict__ ehalf, const int* __restrict__ ei,
                            int E, float* __restrict__ agg)
{
    long idx = (long)blockIdx.x * blockDim.x + threadIdx.x;
    long total = (long)2 * E * 64;
    if (idx >= total) return;
    int d  = (int)(idx & 63);
    int e2 = (int)(idx >> 6);
    int r  = (e2 < E) ? ei[e2 + E] : ei[e2 - E];
    int es = (e2 < E) ? e2 : e2 - E;
    atomicAdd(&agg[(size_t)r * 64 + d], ehalf[(size_t)es * 64 + d]);
}

// ------------- comb = LN(concat(node, agg-node)) over 128 features -----------
__global__ void comb_ln(const float* __restrict__ node, const float* __restrict__ agg,
                        const float* __restrict__ g, const float* __restrict__ b,
                        float* __restrict__ comb, int Nn)
{
    int n = blockIdx.x * blockDim.x + threadIdx.x;
    if (n >= Nn) return;
    const float* nd = node + (size_t)n * 64;
    const float* ag = agg  + (size_t)n * 64;
    float s = 0.f, s2 = 0.f;
    for (int i = 0; i < 128; ++i) {
        float c = (i < 64) ? nd[i] : (ag[i - 64] - nd[i - 64]);
        s += c; s2 += c * c;
    }
    float mean = s * (1.0f / 128.0f);
    float var  = s2 * (1.0f / 128.0f) - mean * mean;
    float inv  = rsqrtf(fmaxf(var, 0.0f) + 1e-5f);
    float* out = comb + (size_t)n * 128;
    for (int i = 0; i < 128; ++i) {
        float c = (i < 64) ? nd[i] : (ag[i - 64] - nd[i - 64]);
        out[i] = (c - mean) * inv * g[i] + b[i];
    }
}

// ------------- per (edge, head) score + atomic segment max -------------------
__global__ void score_max(const float* __restrict__ q, const float* __restrict__ k,
                          const int* __restrict__ ei, int E,
                          float* __restrict__ scores, unsigned* __restrict__ mkeys)
{
    int idx = blockIdx.x * blockDim.x + threadIdx.x;
    if (idx >= 2 * E * 4) return;
    int h  = idx & 3;
    int e2 = idx >> 2;
    int r  = (e2 < E) ? ei[e2 + E] : ei[e2 - E];
    int s  = ei[e2];
    const float* qp = q + (size_t)r * 256 + h * 64;
    const float* kp = k + (size_t)s * 256 + h * 64;
    float acc = 0.f;
    for (int j = 0; j < 64; ++j) acc += qp[j] * kp[j];
    acc *= 0.125f;                                       // 1/sqrt(64)
    scores[idx] = acc;
    atomicMax(&mkeys[r * 4 + h], fkey(acc));
}

// ------------- ex = exp(score - max), atomic segment sum (denom) -------------
__global__ void exp_denom(const int* __restrict__ ei, int E,
                          float* __restrict__ scores, const unsigned* __restrict__ mkeys,
                          float* __restrict__ denom)
{
    int idx = blockIdx.x * blockDim.x + threadIdx.x;
    if (idx >= 2 * E * 4) return;
    int h  = idx & 3;
    int e2 = idx >> 2;
    int r  = (e2 < E) ? ei[e2 + E] : ei[e2 - E];
    float m = funkey(mkeys[r * 4 + h]);
    float e = expf(scores[idx] - m);
    scores[idx] = e;
    atomicAdd(&denom[r * 4 + h], e);
}

// ------------- alpha = ex / (denom[recv] + eps), in place --------------------
__global__ void alpha_fin(const int* __restrict__ ei, int E,
                          float* __restrict__ ex, const float* __restrict__ denom)
{
    int idx = blockIdx.x * blockDim.x + threadIdx.x;
    if (idx >= 2 * E * 4) return;
    int h  = idx & 3;
    int e2 = idx >> 2;
    int r  = (e2 < E) ? ei[e2 + E] : ei[e2 - E];
    ex[idx] = ex[idx] / (denom[(size_t)r * 4 + h] + 1e-16f);
}

// ------------- attn[recv] += alpha * v[send] (scatter over 256 feats) --------
__global__ void attn_scatter(const float* __restrict__ alpha,
                             const float* __restrict__ v, const int* __restrict__ ei,
                             int E, float* __restrict__ attn)
{
    long idx = (long)blockIdx.x * blockDim.x + threadIdx.x;
    long total = (long)2 * E * 256;
    if (idx >= total) return;
    int f  = (int)(idx & 255);
    int e2 = (int)(idx >> 8);
    int h  = f >> 6;
    int r  = (e2 < E) ? ei[e2 + E] : ei[e2 - E];
    int s  = ei[e2];
    float w = alpha[(size_t)e2 * 4 + h];
    atomicAdd(&attn[(size_t)r * 256 + f], w * v[(size_t)s * 256 + f]);
}

// ------------- outn = LN(attn + skip) over 256 features ----------------------
__global__ void outn_ln(const float* __restrict__ attn, const float* __restrict__ skip,
                        const float* __restrict__ g, const float* __restrict__ b,
                        float* __restrict__ outn, int Nn)
{
    int n = blockIdx.x * blockDim.x + threadIdx.x;
    if (n >= Nn) return;
    const float* ap = attn + (size_t)n * 256;
    const float* sp = skip + (size_t)n * 256;
    float s = 0.f, s2 = 0.f;
    for (int i = 0; i < 256; ++i) { float c = ap[i] + sp[i]; s += c; s2 += c * c; }
    float mean = s * (1.0f / 256.0f);
    float var  = s2 * (1.0f / 256.0f) - mean * mean;
    float inv  = rsqrtf(fmaxf(var, 0.0f) + 1e-5f);
    float* out = outn + (size_t)n * 256;
    for (int i = 0; i < 256; ++i) {
        float c = ap[i] + sp[i];
        out[i] = (c - mean) * inv * g[i] + b[i];
    }
}

// ------------- classifier input gather: concat(node[start], node[end]) -------
__global__ void clf_gather(const float* __restrict__ node_new, const int* __restrict__ ei,
                           int E, float* __restrict__ clf)
{
    long idx = (long)blockIdx.x * blockDim.x + threadIdx.x;
    long total = (long)E * 128;
    if (idx >= total) return;
    int f = (int)(idx & 127);
    int e = (int)(idx >> 7);
    int a  = ei[e];          // start = edge_index[0]
    int bb = ei[E + e];      // end   = edge_index[1]
    clf[idx] = (f < 64) ? node_new[(size_t)a * 64 + f]
                        : node_new[(size_t)bb * 64 + (f - 64)];
}

// ------------- final 64->1 projection ----------------------------------------
__global__ void final_proj(const float* __restrict__ h1c, const float* __restrict__ w2,
                           const float* __restrict__ b2, float* __restrict__ out, int E)
{
    int e = blockIdx.x * blockDim.x + threadIdx.x;
    if (e >= E) return;
    const float* hp = h1c + (size_t)e * 64;
    float acc = b2[0];
    for (int j = 0; j < 64; ++j) acc += hp[j] * w2[j];
    out[e] = acc;
}

// ---------------------------------------------------------------------------
extern "C" void kernel_launch(void* const* d_in, const int* in_sizes, int n_in,
                              void* d_out, int out_size, void* d_ws, size_t ws_size,
                              hipStream_t stream)
{
    const float* x         = (const float*)d_in[0];
    const float* edge_attr = (const float*)d_in[1];
    const float* ne_w1 = (const float*)d_in[2];  const float* ne_b1 = (const float*)d_in[3];
    const float* ne_w2 = (const float*)d_in[4];  const float* ne_b2 = (const float*)d_in[5];
    const float* ee_w1 = (const float*)d_in[6];  const float* ee_b1 = (const float*)d_in[7];
    const float* ee_w2 = (const float*)d_in[8];  const float* ee_b2 = (const float*)d_in[9];
    const float* lnc_g = (const float*)d_in[10]; const float* lnc_b = (const float*)d_in[11];
    const float* tq_w  = (const float*)d_in[12]; const float* tq_b  = (const float*)d_in[13];
    const float* tk_w  = (const float*)d_in[14]; const float* tk_b  = (const float*)d_in[15];
    const float* tv_w  = (const float*)d_in[16]; const float* tv_b  = (const float*)d_in[17];
    const float* ts_w  = (const float*)d_in[18]; const float* ts_b  = (const float*)d_in[19];
    const float* lnt_g = (const float*)d_in[20]; const float* lnt_b = (const float*)d_in[21];
    const float* pn_w  = (const float*)d_in[22]; const float* pn_b  = (const float*)d_in[23];
    // d_in[24], d_in[25] = pe_w, pe_b : dead for the output (1 iteration) -> unused
    const float* ec_w1 = (const float*)d_in[26]; const float* ec_b1 = (const float*)d_in[27];
    const float* ec_w2 = (const float*)d_in[28]; const float* ec_b2 = (const float*)d_in[29];
    const int*   ei    = (const int*)d_in[30];   // [2,E] flat: ei[0..E)=row0, ei[E..2E)=row1

    const int Nn = in_sizes[0] / 6;      // 50000
    const int Ee = in_sizes[1] / 4;      // 200000

    // ---- workspace overlay: 6 lanes of S floats (S = max(N*256, E*64)) ----
    float* ws = (float*)d_ws;
    size_t S = (size_t)Nn * 256;
    if ((size_t)Ee * 64 > S) S = (size_t)Ee * 64;
    float* misc = ws;
    float* L1 = ws + 1 * S;   // e_half -> k -> outn -> h1c
    float* L2 = ws + 2 * S;   // q -> clf (spans L2+L3)
    float* L3 = ws + 3 * S;   // v
    float* L4 = ws + 4 * S;   // eh1 -> skip(s)
    float* L5 = ws + 5 * S;   // attn
    float* nodeB = misc;                              // [N,64]  (long-lived)
    float* slot2 = misc + (size_t)Nn * 64;            // nh1 -> agg -> node_new
    float* combB = misc + (size_t)Nn * 128;           // comb [N,128]; later ex/mkeys/denom
    float*    exB    = combB;                         // [2E,4]
    unsigned* mkeysB = (unsigned*)(combB + (size_t)2 * Ee * 4);     // [N,4]
    float*    denomB = combB + (size_t)2 * Ee * 4 + (size_t)Nn * 4; // [N,4]

    auto gemm = [&](const float* A, const float* W, const float* bias, const float* resid,
                    float* Dst, int M, int K, int Nc, int relu) {
        dim3 blk(32, 4, 1);
        dim3 grd(Nc / 16, (M / 16 + 3) / 4, 1);
        wmma_gemm<<<grd, blk, 0, stream>>>(A, W, bias, resid, Dst, M, K, Nc, relu);
    };
    auto nb = [](long total) { return dim3((unsigned)((total + 255) / 256)); };

    // 1-2: node encoder  x[N,6] -> relu -> [N,64] -> relu -> node[N,64]
    gemm(x, ne_w1, ne_b1, nullptr, slot2, Nn, 6, 64, 1);
    gemm(slot2, ne_w2, ne_b2, nullptr, nodeB, Nn, 64, 64, 1);

    // 3: edge encoder on E rows only (bidirectional rows are exact duplicates)
    gemm(edge_attr, ee_w1, ee_b1, nullptr, L4, Ee, 4, 64, 1);
    gemm(L4, ee_w2, ee_b2, nullptr, L1, Ee, 64, 64, 1);          // e_half [E,64]

    // 4: agg = segment_sum(edge, recv)
    hipMemsetAsync(slot2, 0, (size_t)Nn * 64 * sizeof(float), stream);
    scatter_agg<<<nb((long)2 * Ee * 64), 256, 0, stream>>>(L1, ei, Ee, slot2);

    // 5: comb = LN(concat(node, agg-node))
    comb_ln<<<nb(Nn), 256, 0, stream>>>(nodeB, slot2, lnc_g, lnc_b, combB, Nn);

    // 6: q,k,v,skip projections [N,128] x [128,256]
    gemm(combB, tq_w, tq_b, nullptr, L2, Nn, 128, 256, 0);       // q
    gemm(combB, tk_w, tk_b, nullptr, L1, Nn, 128, 256, 0);       // k
    gemm(combB, tv_w, tv_b, nullptr, L3, Nn, 128, 256, 0);       // v
    gemm(combB, ts_w, ts_b, nullptr, L4, Nn, 128, 256, 0);       // skip

    // 7: attention softmax over incoming edges per destination
    hipMemsetAsync(mkeysB, 0, (size_t)Nn * 4 * sizeof(unsigned), stream);
    hipMemsetAsync(denomB, 0, (size_t)Nn * 4 * sizeof(float), stream);
    score_max<<<nb((long)2 * Ee * 4), 256, 0, stream>>>(L2, L1, ei, Ee, exB, mkeysB);
    exp_denom<<<nb((long)2 * Ee * 4), 256, 0, stream>>>(ei, Ee, exB, mkeysB, denomB);
    alpha_fin<<<nb((long)2 * Ee * 4), 256, 0, stream>>>(ei, Ee, exB, denomB);

    // 8: attn = segment_sum(alpha * v[send])
    hipMemsetAsync(L5, 0, (size_t)Nn * 256 * sizeof(float), stream);
    attn_scatter<<<nb((long)2 * Ee * 256), 256, 0, stream>>>(exB, L3, ei, Ee, L5);

    // 9: outn = LN(attn + skip)
    outn_ln<<<nb(Nn), 256, 0, stream>>>(L5, L4, lnt_g, lnt_b, L1, Nn);

    // 10: node = outn @ pn_w + pn_b + x0
    gemm(L1, pn_w, pn_b, nodeB, slot2, Nn, 256, 64, 0);          // node_new in slot2

    // 11-13: classifier over original E edges
    clf_gather<<<nb((long)Ee * 128), 256, 0, stream>>>(slot2, ei, Ee, L2);  // clf spans L2+L3
    gemm(L2, ec_w1, ec_b1, nullptr, L1, Ee, 128, 64, 1);         // h1c [E,64]
    final_proj<<<nb(Ee), 256, 0, stream>>>(L1, ec_w2, ec_b2, (float*)d_out, Ee);
}
